// PointnetSAModuleVotes_35802847379561
// MI455X (gfx1250) — matrix operand order repaired and checked by reading, not compiled, not run on hardware
//
#include <hip/hip_runtime.h>
#include <hip/hip_bf16.h>

typedef __attribute__((ext_vector_type(16))) __bf16 v16bf;
typedef __attribute__((ext_vector_type(8)))  __bf16 v8bf;
typedef __attribute__((ext_vector_type(8)))  float  v8f;

#define B_    4
#define N_    16384
#define C_    16
#define S_    1024
#define NS_   64
#define P_    (B_*S_*NS_)   /* 262144 */
#define RAD_  0.4f
#define RAD2_ (RAD_*RAD_)
#define EPS_  1e-5f

// ---------------------------------------------------------------------------
// Furthest point sampling: one block per batch, 1024 threads.
// Whole batch point cloud (192 KB) staged once into CDNA5's big WGP LDS;
// the 1024 serial steps then run on ds_load + VALU with register-resident
// min-distances. Only 2 workgroup barriers per step.
// ---------------------------------------------------------------------------
__global__ __launch_bounds__(1024)
void fps_kernel(const float* __restrict__ xyz, float* __restrict__ newxyz,
                int* __restrict__ inds, long long* __restrict__ inds64)
{
    int b = blockIdx.x, tid = threadIdx.x;
    const float* px = xyz + (size_t)b * N_ * 3;

    __shared__ float spts[N_ * 3];          // 192 KB
    __shared__ float cx, cy, cz;
    __shared__ float redV[32];
    __shared__ int   redI[32];

    // one-time staging, vectorized
    {
        const float4* s4 = (const float4*)px;
        float4* d4 = (float4*)spts;
        for (int i = tid; i < N_ * 3 / 4; i += 1024) d4[i] = s4[i];
    }

    float dist[16];
#pragma unroll
    for (int j = 0; j < 16; ++j) dist[j] = 1e10f;

    if (tid == 0) {                          // step 0: centroid = point 0
        inds[b * S_] = 0;
        inds64[b * S_] = 0;
        cx = spts[0]; cy = spts[1]; cz = spts[2];
        newxyz[(size_t)b * S_ * 3 + 0] = spts[0];
        newxyz[(size_t)b * S_ * 3 + 1] = spts[1];
        newxyz[(size_t)b * S_ * 3 + 2] = spts[2];
    }
    __syncthreads();

    for (int it = 0; it < S_; ++it) {
        float bx = cx, by = cy, bz = cz;
        float bestV = -1.f; int bestI = 0;
#pragma unroll
        for (int j = 0; j < 16; ++j) {
            int p = tid + j * 1024;
            float dx = spts[p * 3] - bx, dy = spts[p * 3 + 1] - by, dz = spts[p * 3 + 2] - bz;
            float d = dx * dx + dy * dy + dz * dz;
            float nd = fminf(dist[j], d);
            dist[j] = nd;
            if (nd > bestV) { bestV = nd; bestI = p; }
        }
        // wave32 argmax (prefer smaller index on tie, like jnp.argmax)
#pragma unroll
        for (int m = 1; m < 32; m <<= 1) {
            float ov = __shfl_xor(bestV, m);
            int   oi = __shfl_xor(bestI, m);
            if (ov > bestV || (ov == bestV && oi < bestI)) { bestV = ov; bestI = oi; }
        }
        if ((tid & 31) == 0) { redV[tid >> 5] = bestV; redI[tid >> 5] = bestI; }
        __syncthreads();
        if (tid < 32) {
            bestV = redV[tid]; bestI = redI[tid];
#pragma unroll
            for (int m = 1; m < 32; m <<= 1) {
                float ov = __shfl_xor(bestV, m);
                int   oi = __shfl_xor(bestI, m);
                if (ov > bestV || (ov == bestV && oi < bestI)) { bestV = ov; bestI = oi; }
            }
            if (tid == 0 && it + 1 < S_) {   // broadcast next centroid
                int f = bestI;
                inds[b * S_ + it + 1] = f;
                inds64[b * S_ + it + 1] = (long long)f;
                float x = spts[f * 3], y = spts[f * 3 + 1], z = spts[f * 3 + 2];
                cx = x; cy = y; cz = z;
                newxyz[(size_t)(b * S_ + it + 1) * 3 + 0] = x;
                newxyz[(size_t)(b * S_ + it + 1) * 3 + 1] = y;
                newxyz[(size_t)(b * S_ + it + 1) * 3 + 2] = z;
            }
        }
        __syncthreads();
    }
}

// ---------------------------------------------------------------------------
// Ball query: one wave32 per center; ordered append via ballot + popcount.
// ---------------------------------------------------------------------------
__global__ __launch_bounds__(256)
void ballquery_kernel(const float* __restrict__ xyz, const float* __restrict__ newxyz,
                      const int* __restrict__ inds, int* __restrict__ idxbuf)
{
    int lane = threadIdx.x & 31;
    int center = blockIdx.x * 8 + (threadIdx.x >> 5);   // 0..B*S-1
    int b = center / S_;
    const float* px = xyz + (size_t)b * N_ * 3;
    float cx = newxyz[center * 3], cy = newxyz[center * 3 + 1], cz = newxyz[center * 3 + 2];
    int count = 0;
    int first = inds[center];               // fallback (center is always within radius)
    int* out = idxbuf + (size_t)center * NS_;

    for (int base = 0; base < N_; base += 32) {
        int p = base + lane;
        float dx = px[p * 3] - cx, dy = px[p * 3 + 1] - cy, dz = px[p * 3 + 2] - cz;
        bool pred = (dx * dx + dy * dy + dz * dz) < RAD2_;
        unsigned mask = (unsigned)__ballot(pred);
        if (mask) {
            if (count == 0) first = base + (__ffs(mask) - 1);
            int before = __popc(mask & ((1u << lane) - 1u));
            int pos = count + before;
            if (pred && pos < NS_) out[pos] = p;
            count += __popc(mask);
            if (count >= NS_) break;
        }
    }
    int c = count < NS_ ? count : NS_;
    for (int j = c + lane; j < NS_; j += 32) out[j] = first;
}

// ---------------------------------------------------------------------------
// Gather + normalize + concat -> X bf16, pixel-major [p][32] (19 real ch, pad 0)
// ---------------------------------------------------------------------------
__global__ __launch_bounds__(256)
void build_kernel(const float* __restrict__ xyz, const float* __restrict__ feat,
                  const float* __restrict__ newxyz, const int* __restrict__ idxbuf,
                  __bf16* __restrict__ X)
{
    int p = blockIdx.x * 256 + threadIdx.x;    // < P_
    int bs = p / NS_;
    int b = bs / S_;
    int pi = idxbuf[p];
    const float* pnt = xyz + ((size_t)b * N_ + pi) * 3;
    const float inv = 1.f / RAD_;
    __bf16* xp = X + (size_t)p * 32;
    xp[0] = (__bf16)((pnt[0] - newxyz[bs * 3 + 0]) * inv);
    xp[1] = (__bf16)((pnt[1] - newxyz[bs * 3 + 1]) * inv);
    xp[2] = (__bf16)((pnt[2] - newxyz[bs * 3 + 2]) * inv);
    const float* fb = feat + (size_t)b * C_ * N_;
#pragma unroll
    for (int c = 0; c < C_; ++c) xp[3 + c] = (__bf16)fb[c * N_ + pi];
#pragma unroll
    for (int c = 19; c < 32; ++c) xp[c] = (__bf16)0.f;
}

// ---------------------------------------------------------------------------
// Weight prep: f32 [M][K] -> bf16 [M][KPAD] zero-padded
// ---------------------------------------------------------------------------
__global__ void wprep_kernel(const float* __restrict__ W, __bf16* __restrict__ Wp,
                             int M, int K, int KPAD)
{
    int t = blockIdx.x * 256 + threadIdx.x;
    if (t >= M * KPAD) return;
    int o = t / KPAD, c = t % KPAD;
    Wp[t] = (c < K) ? (__bf16)W[o * K + c] : (__bf16)0.f;
}

// ---------------------------------------------------------------------------
// WMMA bf16 GEMM with optional fused input BN+ReLU:
//   Y[p][o] = sum_c Wp[o][c] * act(X[p][c])
//   act(x) = FUSE ? relu(x*scale[c]+bias[c]) : x
// Each wave computes a 16(M) x 64(N) strip; A-fragment reused across 4 WMMAs.
// Fusing BN into the B-fragment load removes an entire 67MB/layer round trip.
// ---------------------------------------------------------------------------
template <int M, int KPAD, bool FUSE>
__global__ __launch_bounds__(256)
void gemm_wmma(const __bf16* __restrict__ X, const __bf16* __restrict__ Wp,
               __bf16* __restrict__ Y,
               const float* __restrict__ scale, const float* __restrict__ bias)
{
    const int NT = 4;                         // N-tiles per wave
    int lane = threadIdx.x & 31;
    int w    = threadIdx.x >> 5;
    int tile = blockIdx.x * 8 + w;
    const int MT = M / 16;
    int mt = tile % MT, ptg = tile / MT;
    int p0 = ptg * (16 * NT), o0 = mt * 16;
    int hi = lane >> 4, ln = lane & 15;

    v8f acc[NT] = {};
#pragma unroll
    for (int kt = 0; kt < KPAD / 32; ++kt) {
        int k0 = kt * 32;
        // A fragment: row o0+ln; 16 consecutive K values k0+hi*16 ..+15
        const __bf16* ap = Wp + (size_t)(o0 + ln) * KPAD + k0 + hi * 16;
        v8bf alo = *(const v8bf*)ap;
        v8bf ahi = *(const v8bf*)(ap + 8);
        v16bf a;
#pragma unroll
        for (int i = 0; i < 8; ++i) { a[i] = alo[i]; a[i + 8] = ahi[i]; }

        // per-channel BN coefficients for this fragment's 16 channels
        float scv[16], biv[16];
        if (FUSE) {
            const float4* sp = (const float4*)(scale + k0 + hi * 16);
            const float4* bp4 = (const float4*)(bias + k0 + hi * 16);
#pragma unroll
            for (int i = 0; i < 4; ++i) {
                float4 s4 = sp[i], b4 = bp4[i];
                scv[4*i+0]=s4.x; scv[4*i+1]=s4.y; scv[4*i+2]=s4.z; scv[4*i+3]=s4.w;
                biv[4*i+0]=b4.x; biv[4*i+1]=b4.y; biv[4*i+2]=b4.z; biv[4*i+3]=b4.w;
            }
        }
#pragma unroll
        for (int j = 0; j < NT; ++j) {
            // B fragment: column p0+j*16+ln; 16 consecutive K values
            const __bf16* bp = X + (size_t)(p0 + j * 16 + ln) * KPAD + k0 + hi * 16;
            __builtin_prefetch((const char*)bp + 32768, 0, 1);
            v16bf bm = *(const v16bf*)bp;
            if (FUSE) {
#pragma unroll
                for (int i = 0; i < 16; ++i)
                    bm[i] = (__bf16)fmaxf(0.f, (float)bm[i] * scv[i] + biv[i]);
            }
            acc[j] = __builtin_amdgcn_wmma_f32_16x16x32_bf16(false, a, false, bm,
                                                             (short)0, acc[j], false, false);
        }
    }

    // D layout: VGPR r -> o = o0 + hi*8 + r, column p = p0 + j*16 + ln
#pragma unroll
    for (int j = 0; j < NT; ++j) {
        v8bf yb;
#pragma unroll
        for (int r = 0; r < 8; ++r) yb[r] = (__bf16)acc[j][r];
        *(v8bf*)(Y + (size_t)(p0 + j * 16 + ln) * M + o0 + hi * 8) = yb;
    }
}

// ---------------------------------------------------------------------------
// Per-channel sum / sum-of-squares over all P pixels (coalesced re-read of Y).
// 128 blocks -> only 128*M atomics total.
// ---------------------------------------------------------------------------
template <int M>
__global__ __launch_bounds__(256)
void redstats_kernel(const __bf16* __restrict__ Y, float* __restrict__ sum,
                     float* __restrict__ sumsq)
{
    const int R = 256 / M;                    // pixel rows handled in parallel
    const int CHUNK = P_ / 128;               // 2048 pixels per block
    int c = threadIdx.x % M;
    int r = threadIdx.x / M;
    size_t p0 = (size_t)blockIdx.x * CHUNK;
    float s = 0.f, q = 0.f;
    for (int j = r; j < CHUNK; j += R) {
        float y = (float)Y[(p0 + j) * M + c];
        s += y; q += y * y;
    }
    __shared__ float ls[256], lq[256];
    ls[threadIdx.x] = s; lq[threadIdx.x] = q;
    __syncthreads();
    if (r == 0) {
#pragma unroll
        for (int k = 1; k < R; ++k) { s += ls[k * M + c]; q += lq[k * M + c]; }
        atomicAdd(&sum[c], s);
        atomicAdd(&sumsq[c], q);
    }
}

// ---------------------------------------------------------------------------
// BN finalize: per-channel scale/bias from batch statistics.
// ---------------------------------------------------------------------------
__global__ void stats_kernel(const float* __restrict__ sum, const float* __restrict__ sumsq,
                             const float* __restrict__ g, const float* __restrict__ bb,
                             float* __restrict__ scale, float* __restrict__ bias, int M)
{
    int o = threadIdx.x;
    if (o >= M) return;
    const float inv = 1.f / (float)P_;
    float mean = sum[o] * inv;
    float var  = sumsq[o] * inv - mean * mean;
    float sc = g[o] * rsqrtf(var + EPS_);
    scale[o] = sc;
    bias[o]  = bb[o] - mean * sc;
}

// ---------------------------------------------------------------------------
// Final: BN3 + ReLU + max-pool over 64 neighbors -> new_features (B,128,S).
// One block per (b,s): 128 threads = 128 channels, every k-step reads a
// contiguous 256B line of Y.
// ---------------------------------------------------------------------------
__global__ __launch_bounds__(128)
void pool_kernel(const __bf16* __restrict__ Y, const float* __restrict__ scale,
                 const float* __restrict__ bias, float* __restrict__ outF)
{
    int bs = blockIdx.x;                 // 0..B*S-1
    int o  = threadIdx.x;                // 0..127
    const __bf16* yp = Y + (size_t)bs * NS_ * 128 + o;
    float sc = scale[o], bi = bias[o];
    float m = 0.f;                       // max(relu(v)) == max(0, max(v))
#pragma unroll 8
    for (int k = 0; k < NS_; ++k)
        m = fmaxf(m, (float)yp[k * 128] * sc + bi);
    int b = bs / S_, s = bs % S_;
    outF[((size_t)b * 128 + o) * S_ + s] = m;
}

// ---------------------------------------------------------------------------
extern "C" void kernel_launch(void* const* d_in, const int* in_sizes, int n_in,
                              void* d_out, int out_size, void* d_ws, size_t ws_size,
                              hipStream_t stream)
{
    (void)in_sizes; (void)n_in; (void)out_size; (void)ws_size;
    const float* xyz  = (const float*)d_in[0];
    const float* feat = (const float*)d_in[1];
    const float* W1 = (const float*)d_in[2];
    const float* g1 = (const float*)d_in[3];
    const float* b1 = (const float*)d_in[4];
    const float* W2 = (const float*)d_in[5];
    const float* g2 = (const float*)d_in[6];
    const float* b2 = (const float*)d_in[7];
    const float* W3 = (const float*)d_in[8];
    const float* g3 = (const float*)d_in[9];
    const float* b3 = (const float*)d_in[10];

    float* outF    = (float*)d_out;
    float* newxyz  = outF;                       // (B,S,3) = 12288 floats
    float* newfeat = outF + B_ * S_ * 3;         // (B,128,S) = 524288 floats
    long long* inds64 = (long long*)(newfeat + B_ * 128 * S_);

    char* ws = (char*)d_ws;
    int*    inds   = (int*)ws;                          // 16 KB
    int*    idxbuf = (int*)(ws + 16384);                // 1 MB
    float*  stats  = (float*)(ws + 16384 + 1048576);    // 2 KB
    float*  sum = stats, *sumsq = stats + 128, *scale = stats + 256, *bias = stats + 384;
    __bf16* W1p = (__bf16*)(ws + 1067008);
    __bf16* W2p = (__bf16*)(ws + 1071104);
    __bf16* W3p = (__bf16*)(ws + 1079296);
    __bf16* X   = (__bf16*)(ws + 1095680);              // P*32 bf16 = 16.7 MB
    __bf16* Ya  = (__bf16*)(ws + 17872896);             // P*128 bf16 = 67 MB (L1/L3 out)
    __bf16* Yb  = (__bf16*)(ws + 84981760);             // P*64  bf16 = 33.5 MB (L2 out)

    fps_kernel<<<B_, 1024, 0, stream>>>(xyz, newxyz, inds, inds64);
    ballquery_kernel<<<(B_ * S_) / 8, 256, 0, stream>>>(xyz, newxyz, inds, idxbuf);
    build_kernel<<<P_ / 256, 256, 0, stream>>>(xyz, feat, newxyz, idxbuf, X);
    wprep_kernel<<<(64 * 32 + 255) / 256, 256, 0, stream>>>(W1, W1p, 64, 19, 32);
    wprep_kernel<<<(64 * 64 + 255) / 256, 256, 0, stream>>>(W2, W2p, 64, 64, 64);
    wprep_kernel<<<(128 * 64 + 255) / 256, 256, 0, stream>>>(W3, W3p, 128, 64, 64);

    // layer 1: 19(->32) -> 64, raw input
    hipMemsetAsync(sum, 0, 256 * sizeof(float), stream);
    gemm_wmma<64, 32, false><<<(P_ / 64) * 4 / 8, 256, 0, stream>>>(X, W1p, Ya, nullptr, nullptr);
    redstats_kernel<64><<<128, 256, 0, stream>>>(Ya, sum, sumsq);
    stats_kernel<<<1, 128, 0, stream>>>(sum, sumsq, g1, b1, scale, bias, 64);

    // layer 2: 64 -> 64, BN1+ReLU fused into the B-fragment load
    hipMemsetAsync(sum, 0, 256 * sizeof(float), stream);
    gemm_wmma<64, 64, true><<<(P_ / 64) * 4 / 8, 256, 0, stream>>>(Ya, W2p, Yb, scale, bias);
    redstats_kernel<64><<<128, 256, 0, stream>>>(Yb, sum, sumsq);
    stats_kernel<<<1, 128, 0, stream>>>(sum, sumsq, g2, b2, scale, bias, 64);

    // layer 3: 64 -> 128, BN2+ReLU fused; then BN3+ReLU+maxpool fused
    hipMemsetAsync(sum, 0, 256 * sizeof(float), stream);
    gemm_wmma<128, 64, true><<<(P_ / 64) * 8 / 8, 256, 0, stream>>>(Yb, W3p, Ya, scale, bias);
    redstats_kernel<128><<<128, 256, 0, stream>>>(Ya, sum, sumsq);
    stats_kernel<<<1, 128, 0, stream>>>(sum, sumsq, g3, b3, scale, bias, 128);
    pool_kernel<<<B_ * S_, 128, 0, stream>>>(Ya, scale, bias, newfeat);
}
